// KNN_Predict_21766894256654
// MI455X (gfx1250) — matrix operand order repaired and checked by reading, not compiled, not run on hardware
//
#include <hip/hip_runtime.h>
#include <hip/hip_bf16.h>

// ---------------- problem constants (from reference) ----------------
#define N_Q    1024
#define D_IN   512
#define D_H    256
#define N_REF  200000
#define D_OUT  128
#define KNN    1000
#define NBINS  1024

typedef __attribute__((ext_vector_type(16))) _Float16 v16h;
typedef __attribute__((ext_vector_type(8)))  _Float16 v8h;
typedef __attribute__((ext_vector_type(8)))  float    v8f;

union V16 { v16h v; v8h h8[2]; };

// ---------------- workspace layout (bytes, all 256B aligned) ----------------
#define OFF_H     0u                       // h_f16   : 1024*256*2   = 524288
#define OFF_R     524288u                  // r_f16   : 200000*256*2 = 102400000
#define OFF_HIST  102924288u               // hist    : 1024*1024*4  = 4194304
#define OFF_THR   107118592u               // thresh  : 1024*4
#define OFF_CNT   107122688u               // cnt     : 1024*4
#define OFF_LIST  107126784u               // list    : 1024*1000*4  = 4096000

// =====================================================================
// Kernel 1: h = normalize_rows(x @ W_enc), store as f16 (row-major)
// =====================================================================
__global__ void encode_kernel(const float* __restrict__ x,
                              const float* __restrict__ W,
                              _Float16* __restrict__ hq) {
    __shared__ float xs[D_IN];
    __shared__ float red[256];
    const int i = blockIdx.x, t = threadIdx.x;
    xs[t]       = x[(size_t)i * D_IN + t];
    xs[t + 256] = x[(size_t)i * D_IN + 256 + t];
    __syncthreads();
    float acc = 0.f;
    #pragma unroll 8
    for (int k = 0; k < D_IN; ++k)
        acc = fmaf(xs[k], W[(size_t)k * D_H + t], acc);
    red[t] = acc * acc;
    __syncthreads();
    for (int s = 128; s > 0; s >>= 1) {
        if (t < s) red[t] += red[t + s];
        __syncthreads();
    }
    const float inv = rsqrtf(red[0]);
    hq[(size_t)i * D_H + t] = (_Float16)(acc * inv);
}

// =====================================================================
// Kernel 2: r_f16 = normalize_rows(ref_x), one block per reference row
// =====================================================================
__global__ void refnorm_kernel(const float* __restrict__ rx,
                               _Float16* __restrict__ rq) {
    __shared__ float red[256];
    const int n = blockIdx.x, t = threadIdx.x;
    const float v = rx[(size_t)n * D_H + t];
    red[t] = v * v;
    __syncthreads();
    for (int s = 128; s > 0; s >>= 1) {
        if (t < s) red[t] += red[t + s];
        __syncthreads();
    }
    const float inv = rsqrtf(red[0]);
    rq[(size_t)n * D_H + t] = (_Float16)(v * inv);
}

// =====================================================================
// Kernel 3: zero the histogram
// =====================================================================
__global__ void zero_kernel(int* __restrict__ p, int n) {
    for (int i = blockIdx.x * blockDim.x + threadIdx.x; i < n;
         i += gridDim.x * blockDim.x)
        p[i] = 0;
}

// =====================================================================
// Kernel 4/6: WMMA similarity, fused with histogram (PASS 1) or
// threshold selection (PASS 2).
//
// Each wave owns TWO 16-ref tiles (32 refs): both B panels (2 x 16 refs
// x 256 K f16 = 128 VGPRs) stay register-resident while the wave sweeps
// all 64 query groups. Each A tile load feeds two independent WMMA
// accumulator chains (c0/c1), so A-load latency and WMMA RAW latency
// overlap, and r_f16 is streamed from HBM exactly once per pass.
//
// __launch_bounds__(256, 1): the kernel needs ~170 VGPRs to keep both
// B panels register-resident; the default occupancy-driven budget
// spilled them to scratch inside the hot loop.
//
// VGPR layouts per CDNA5 ISA 7.12.2:
//   A 16x32 f16 : lane L -> row M=L&15, K = {b..b+7, b+16..b+23}, b=(L<16?0:8)
//   B 32x16 f16 : lane L -> col N=L&15, K = {b..b+7, b+16..b+23}, b=(L<16?0:8)
//   C 16x16 f32 : lane L, vgpr v -> M = v + (L<16?0:8), N = L&15
// =====================================================================
template <int PASS>
__global__ void __launch_bounds__(256, 1)
sim_kernel(const _Float16* __restrict__ h,
           const _Float16* __restrict__ r,
           int* __restrict__ hist,
           const float* __restrict__ thresh,
           int* __restrict__ cnt,
           int* __restrict__ sel) {
    const int lane = threadIdx.x & 31;
    const int wave = threadIdx.x >> 5;
    const int pair = blockIdx.x * 8 + wave;        // 32 refs per wave
    if (pair >= N_REF / 32) return;                // whole wave exits together
    const int n0     = pair * 32;
    const int ncol0  = n0 + (lane & 15);
    const int ncol1  = ncol0 + 16;
    const int hiHalf = (lane >> 4) & 1;

    // ---- load both B panels: 2 x (16 refs x 256 dims f16) ----
    V16 b0[8], b1[8];
    const _Float16* r0 = r + (size_t)ncol0 * D_H + hiHalf * 8;
    const _Float16* r1 = r + (size_t)ncol1 * D_H + hiHalf * 8;
    #pragma unroll
    for (int k = 0; k < 8; ++k) {
        b0[k].h8[0] = *(const v8h*)(r0 + k * 32);
        b0[k].h8[1] = *(const v8h*)(r0 + k * 32 + 16);
        b1[k].h8[0] = *(const v8h*)(r1 + k * 32);
        b1[k].h8[1] = *(const v8h*)(r1 + k * 32 + 16);
    }

    // ---- sweep all 64 query groups against the held B panels ----
    for (int qg = 0; qg < N_Q / 16; ++qg) {
        v8f c0 = {}, c1 = {};
        const _Float16* hrow =
            h + (size_t)(qg * 16 + (lane & 15)) * D_H + hiHalf * 8;
        #pragma unroll
        for (int k = 0; k < 8; ++k) {
            V16 a;
            a.h8[0] = *(const v8h*)(hrow + k * 32);
            a.h8[1] = *(const v8h*)(hrow + k * 32 + 16);
            c0 = __builtin_amdgcn_wmma_f32_16x16x32_f16(
                     false, a.v, false, b0[k].v, (short)0, c0, false, false);
            c1 = __builtin_amdgcn_wmma_f32_16x16x32_f16(
                     false, a.v, false, b1[k].v, (short)0, c1, false, false);
        }
        #pragma unroll
        for (int v = 0; v < 8; ++v) {
            const int   q  = qg * 16 + v + hiHalf * 8;
            const float s0 = c0[v];
            const float s1 = c1[v];
            if (PASS == 1) {
                int bin0 = (int)((s0 + 1.0f) * (NBINS * 0.5f));
                int bin1 = (int)((s1 + 1.0f) * (NBINS * 0.5f));
                bin0 = bin0 < 0 ? 0 : (bin0 > NBINS - 1 ? NBINS - 1 : bin0);
                bin1 = bin1 < 0 ? 0 : (bin1 > NBINS - 1 ? NBINS - 1 : bin1);
                atomicAdd(&hist[q * NBINS + bin0], 1);
                atomicAdd(&hist[q * NBINS + bin1], 1);
            } else {
                const float t = thresh[q];
                if (s0 >= t) {
                    const int pos = atomicAdd(&cnt[q], 1);
                    if (pos < KNN) sel[q * KNN + pos] = ncol0;
                }
                if (s1 >= t) {
                    const int pos = atomicAdd(&cnt[q], 1);
                    if (pos < KNN) sel[q * KNN + pos] = ncol1;
                }
            }
        }
    }
}

// =====================================================================
// Kernel 5: per-query scan of histogram (top-down) -> threshold value
// =====================================================================
__global__ void scan_kernel(const int* __restrict__ hist,
                            float* __restrict__ thresh,
                            int* __restrict__ cnt) {
    const int q = blockIdx.x * blockDim.x + threadIdx.x;
    if (q >= N_Q) return;
    int cum = 0;
    float tv = -2.0f;                      // accept-all fallback
    for (int b = NBINS - 1; b >= 0; --b) {
        cum += hist[q * NBINS + b];
        if (cum >= KNN) { tv = (float)b * (2.0f / NBINS) - 1.0f; break; }
    }
    thresh[q] = tv;
    cnt[q]    = 0;
}

// =====================================================================
// Kernel 7: gather-mean of ref_y over selected neighbors
// =====================================================================
__global__ void avg_kernel(const int* __restrict__ sel,
                           const int* __restrict__ cnt,
                           const float* __restrict__ ry,
                           float* __restrict__ out) {
    const int q = blockIdx.x, d = threadIdx.x;
    int c = cnt[q];
    if (c > KNN) c = KNN;
    float s = 0.f;
    for (int j = 0; j < c; ++j)
        s += ry[(size_t)sel[q * KNN + j] * D_OUT + d];
    out[(size_t)q * D_OUT + d] = s / (float)(c > 0 ? c : 1);
}

// =====================================================================
extern "C" void kernel_launch(void* const* d_in, const int* in_sizes, int n_in,
                              void* d_out, int out_size, void* d_ws, size_t ws_size,
                              hipStream_t stream) {
    const float* x    = (const float*)d_in[0];   // [1024, 512]
    const float* Wenc = (const float*)d_in[1];   // [512, 256]
    const float* refx = (const float*)d_in[2];   // [200000, 256]
    const float* refy = (const float*)d_in[3];   // [200000, 128]
    float*       out  = (float*)d_out;           // [1024, 128]

    char* ws = (char*)d_ws;
    _Float16* h_f16 = (_Float16*)(ws + OFF_H);
    _Float16* r_f16 = (_Float16*)(ws + OFF_R);
    int*      hist  = (int*)(ws + OFF_HIST);
    float*    thr   = (float*)(ws + OFF_THR);
    int*      cnt   = (int*)(ws + OFF_CNT);
    int*      sel   = (int*)(ws + OFF_LIST);

    // 1) encoder + query normalization -> f16
    encode_kernel<<<N_Q, 256, 0, stream>>>(x, Wenc, h_f16);
    // 2) reference normalization -> f16
    refnorm_kernel<<<N_REF, 256, 0, stream>>>(refx, r_f16);
    // 3) clear histogram
    zero_kernel<<<512, 256, 0, stream>>>(hist, N_Q * NBINS);
    // 4) WMMA pass 1: similarity histogram
    const int simBlocks = (N_REF / 32 + 7) / 8;   // 6250 wave-pairs / 8 waves
    sim_kernel<1><<<simBlocks, 256, 0, stream>>>(h_f16, r_f16, hist, thr, cnt, sel);
    // 5) per-query threshold
    scan_kernel<<<(N_Q + 255) / 256, 256, 0, stream>>>(hist, thr, cnt);
    // 6) WMMA pass 2: select top-1000 indices
    sim_kernel<2><<<simBlocks, 256, 0, stream>>>(h_f16, r_f16, hist, thr, cnt, sel);
    // 7) gather-average neighbor labels
    avg_kernel<<<N_Q, D_OUT, 0, stream>>>(sel, cnt, refy, out);
}
